// Pointnet2MSG_24283745092086
// MI455X (gfx1250) — matrix operand order repaired and checked by reading, not compiled, not run on hardware
//
#include <hip/hip_runtime.h>
#include <hip/hip_bf16.h>
#include <stdint.h>

// ---------------- problem constants (match reference) ----------------
#define B_    2
#define N_    16384
#define S_    4096
#define CIN_  6
#define CIMG_ 32
#define HIMG_ 192
#define WIMG_ 640
#define PC_   96
#define RC_   24
#define NS0_  16
#define NS1_  32
#define M0_   (B_*S_*NS0_)   // 131072
#define M1_   (B_*S_*NS1_)   // 262144
#define MS_   (B_*S_)        // 8192
#define EPSF_ 1e-5f

typedef __attribute__((ext_vector_type(16))) _Float16 v16h;
typedef __attribute__((ext_vector_type(8)))  _Float16 v8h;
typedef __attribute__((ext_vector_type(8)))  float    v8f;
typedef unsigned int u32x4 __attribute__((ext_vector_type(4)));
typedef int          i32x8 __attribute__((ext_vector_type(8)));
typedef int          i32x4 __attribute__((ext_vector_type(4)));

// A-fragment column permutation (involution): within each 32-wide K tile the
// WMMA A layout wants blocks [0-7,16-23] (lanes 0-15) and [8-15,24-31]
// (lanes 16-31) contiguous, i.e. swap 8-wide blocks 1 and 2.
__device__ __forceinline__ int permk(int k) {
  int b = (k >> 3) & 3;
  int pb = (b == 1) ? 2 : ((b == 2) ? 1 : b);
  return (k & ~31) | (pb << 3) | (k & 7);
}

// ---------------- tiny utility kernels ----------------
__global__ void k_zero_f32(float* p, int n) {
  int t = blockIdx.x * blockDim.x + threadIdx.x;
  if (t < n) p[t] = 0.f;
}

// pack f32 [K,Nc] weight into B-fragment layout:
// dst[(kt*Npad + col)*32 + kk] = src[(kt*32+permk(kk)), col], zero padded.
__global__ void k_pack_w(const float* __restrict__ src, _Float16* __restrict__ dst,
                         int K, int Nc, int KT, int Npad) {
  int t = blockIdx.x * blockDim.x + threadIdx.x;
  if (t >= KT * Npad * 32) return;
  int kk  = t & 31;
  int col = (t >> 5) % Npad;
  int kt  = t / (32 * Npad);
  int kg  = kt * 32 + permk(kk);
  float v = (kg < K && col < Nc) ? src[kg * Nc + col] : 0.f;
  dst[t] = (_Float16)v;
}

// ---------------- farthest point sampling ----------------
__global__ void k_fps(const float* __restrict__ xyz, float* __restrict__ dists,
                      int* __restrict__ fps_idx) {
  const int b = blockIdx.x;
  const int tid = threadIdx.x;
  const float* px = xyz + (size_t)b * N_ * 3;
  float* dd = dists + (size_t)b * N_;
  __shared__ float rv[1024];
  __shared__ int   rixx[1024];
  __shared__ int   s_far;
  int far = 0;
  for (int t = 0; t < S_; ++t) {
    if (tid == 0) fps_idx[b * S_ + t] = far;
    const float fx = px[3 * far], fy = px[3 * far + 1], fz = px[3 * far + 2];
    float bv = -1.f; int bi = 0;
    for (int i = tid; i < N_; i += 1024) {
      float dx = px[3 * i] - fx, dy = px[3 * i + 1] - fy, dz = px[3 * i + 2] - fz;
      float d = dx * dx + dy * dy + dz * dz;
      float nd = (t == 0) ? d : fminf(dd[i], d);
      dd[i] = nd;
      if (nd > bv) { bv = nd; bi = i; }
    }
    rv[tid] = bv; rixx[tid] = bi;
    __syncthreads();
    for (int off = 512; off > 0; off >>= 1) {
      if (tid < off) {
        float ov = rv[tid + off]; int oi = rixx[tid + off];
        if (ov > rv[tid] || (ov == rv[tid] && oi < rixx[tid])) { rv[tid] = ov; rixx[tid] = oi; }
      }
      __syncthreads();
    }
    if (tid == 0) s_far = rixx[0];
    __syncthreads();
    far = s_far;
  }
}

__global__ void k_gather(const float* __restrict__ xyz, const float* __restrict__ xy,
                         const int* __restrict__ fps_idx,
                         float* __restrict__ new_xyz, float* __restrict__ li_xy) {
  int t = blockIdx.x * blockDim.x + threadIdx.x;
  if (t >= MS_) return;
  int b = t / S_;
  int i = fps_idx[t];
  const float* p = xyz + ((size_t)b * N_ + i) * 3;
  new_xyz[t * 3 + 0] = p[0];
  new_xyz[t * 3 + 1] = p[1];
  new_xyz[t * 3 + 2] = p[2];
  const float* q = xy + ((size_t)b * N_ + i) * 2;
  li_xy[t * 2 + 0] = q[0];
  li_xy[t * 2 + 1] = q[1];
}

// ---------------- ball query (both radii in one scan) ----------------
__global__ void k_ballquery(const float* __restrict__ xyz, const float* __restrict__ new_xyz,
                            int* __restrict__ idx0, int* __restrict__ idx1) {
  int t = blockIdx.x * blockDim.x + threadIdx.x;
  if (t >= MS_) return;
  int b = t / S_;
  const float qx = new_xyz[t * 3], qy = new_xyz[t * 3 + 1], qz = new_xyz[t * 3 + 2];
  const float* px = xyz + (size_t)b * N_ * 3;
  int* o0 = idx0 + (size_t)t * NS0_;
  int* o1 = idx1 + (size_t)t * NS1_;
  int c0 = 0, c1 = 0, f0 = -1, f1 = -1;
  for (int i = 0; i < N_; ++i) {
    if (i + 256 < N_) __builtin_prefetch(px + 3 * (i + 256), 0, 1);
    float dx = px[3 * i] - qx, dy = px[3 * i + 1] - qy, dz = px[3 * i + 2] - qz;
    float d2 = dx * dx + dy * dy + dz * dz;
    if (c1 < NS1_ && d2 < 1.0f)  { if (c1 == 0) f1 = i; o1[c1++] = i; }
    if (c0 < NS0_ && d2 < 0.25f) { if (c0 == 0) f0 = i; o0[c0++] = i; }
    if (c0 == NS0_ && c1 == NS1_) break;
  }
  int fill0 = (f0 < 0) ? 0 : f0;
  int fill1 = (f1 < 0) ? 0 : f1;
  for (int k = c0; k < NS0_; ++k) o0[k] = fill0;
  for (int k = c1; k < NS1_; ++k) o1[k] = fill1;
}

// ---------------- build grouped A matrix: [M, 32] f16, perm layout ----------------
__global__ void k_group(const float* __restrict__ xyz, const float* __restrict__ feats,
                        const float* __restrict__ new_xyz, const int* __restrict__ idx,
                        int ns, _Float16* __restrict__ A, int M) {
  int t = blockIdx.x * blockDim.x + threadIdx.x;
  if (t >= M) return;
  int s = (t / ns) % S_;
  int b = t / (ns * S_);
  int i = idx[t];
  const float* p = xyz + ((size_t)b * N_ + i) * 3;
  const float* q = new_xyz + (size_t)(b * S_ + s) * 3;
  _Float16* a = A + (size_t)t * 32;
#pragma unroll
  for (int k = 0; k < 32; ++k) a[k] = (_Float16)0.f;
  a[permk(0)] = (_Float16)(p[0] - q[0]);
  a[permk(1)] = (_Float16)(p[1] - q[1]);
  a[permk(2)] = (_Float16)(p[2] - q[2]);
#pragma unroll
  for (int c = 0; c < CIN_; ++c)
    a[permk(3 + c)] = (_Float16)feats[((size_t)b * CIN_ + c) * N_ + i];
}

// ---------------- WMMA GEMM (templated on tile counts) ----------------
// A f16 [M,KT*32] in perm layout (lda), W f16 in B-fragment layout (LDS-staged
// via TDM when available), Y f16 [M,NT*16] logical layout, optional BN stats.
#define GEMM_WAVES 8
#define GEMM_MT 4
template <int KT, int NT>
__global__ __launch_bounds__(256) void k_gemm_t(
    const _Float16* __restrict__ A, int lda,
    const _Float16* __restrict__ Wt,
    _Float16* __restrict__ Y, int ldy,
    int M, float* __restrict__ stats) {
  constexpr int NPAD = NT * 16;
  constexpr int PANEL_HALVES = KT * 32 * NPAD;
  __shared__ __attribute__((aligned(32))) _Float16 sW[PANEL_HALVES];
  __shared__ float sStat[2 * NPAD];
  const int tid  = threadIdx.x;
  const int lane = tid & 31;
  const int wave = tid >> 5;

  if (stats) for (int i = tid; i < 2 * NPAD; i += blockDim.x) sStat[i] = 0.f;

#if __has_builtin(__builtin_amdgcn_tensor_load_to_lds) && __has_builtin(__builtin_amdgcn_s_wait_tensorcnt)
  // Stage weight panel with the Tensor Data Mover (1-D descriptor, 4B elems).
  if (wave == 0) {
    const unsigned long long ga = (unsigned long long)(uintptr_t)Wt;
    const unsigned ldsoff = (unsigned)(uintptr_t)(void*)sW;  // low 32 bits = LDS byte offset
    constexpr unsigned T4 = (unsigned)(PANEL_HALVES / 2);    // dwords, fits 16-bit tile_dim0
    u32x4 g0;
    g0[0] = 1u;                                             // count=1 (valid user D#)
    g0[1] = ldsoff;                                         // lds_addr
    g0[2] = (unsigned)(ga & 0xFFFFFFFFu);                   // global_addr[31:0]
    g0[3] = (unsigned)((ga >> 32) & 0x01FFFFFFu) | 0x80000000u; // addr[56:32] | type=2
    i32x8 g1;
    g1[0] = (int)(2u << 16);                 // data_size=4B; no multicast/iterate/pad
    g1[1] = (int)((T4 & 0xFFFFu) << 16);     // tensor_dim0[15:0]
    g1[2] = (int)((T4 >> 16) | (1u << 16));  // tensor_dim0[31:16], tensor_dim1=1
    g1[3] = (int)((T4 & 0xFFFFu) << 16);     // tile_dim0 = T4
    g1[4] = 0;                               // tile_dim1=0, tile_dim2=0 (unused)
    g1[5] = (int)T4;                         // tensor_dim0_stride[31:0]
    g1[6] = 0;
    g1[7] = 0;
    i32x4 z4 = {0, 0, 0, 0};
    i32x8 z8 = {0, 0, 0, 0, 0, 0, 0, 0};
    __builtin_amdgcn_tensor_load_to_lds(g0, g1, z4, z4, z8, 0);
    __builtin_amdgcn_s_wait_tensorcnt(0);
  }
#else
  {
    const uint4* src = (const uint4*)Wt;
    uint4* dst = (uint4*)sW;
    for (int i = tid; i < PANEL_HALVES / 8; i += blockDim.x) dst[i] = src[i];
  }
#endif
  __syncthreads();

  const int khalf = lane >> 4;
  const int l15   = lane & 15;
  const int Mtiles = M >> 4;

  for (int mt = 0; mt < GEMM_MT; ++mt) {
    const int mtile = (blockIdx.x * GEMM_WAVES + wave) * GEMM_MT + mt;
    if (mtile >= Mtiles) break;
    const int arow = (mtile << 4) + l15;
    v16h afr[KT];
#pragma unroll
    for (int kt = 0; kt < KT; ++kt)
      afr[kt] = *(const v16h*)(A + (size_t)arow * lda + kt * 32 + (khalf << 4));
#pragma unroll
    for (int nt = 0; nt < NT; ++nt) {
      const int col = nt * 16 + l15;
      v8f acc = {};
#pragma unroll
      for (int kt = 0; kt < KT; ++kt) {
        v16h bf = *(const v16h*)(sW + (((kt * NPAD + col) << 5) + (khalf << 4)));
        acc = __builtin_amdgcn_wmma_f32_16x16x32_f16(
            /*neg_a=*/false, afr[kt], /*neg_b=*/false, bf,
            /*c_mod=*/(short)0, acc, /*reuse_a=*/false, /*reuse_b=*/false);
      }
      const int rbase = (mtile << 4) + (khalf << 3);
#pragma unroll
      for (int j = 0; j < 8; ++j)
        Y[(size_t)(rbase + j) * ldy + col] = (_Float16)acc[j];
      if (stats) {
        float ps = 0.f, pq = 0.f;
#pragma unroll
        for (int j = 0; j < 8; ++j) { float v = acc[j]; ps += v; pq += v * v; }
        atomicAdd(&sStat[col], ps);
        atomicAdd(&sStat[NPAD + col], pq);
      }
    }
  }
  if (stats) {
    __syncthreads();
    for (int c = tid; c < 2 * NPAD; c += blockDim.x) atomicAdd(&stats[c], sStat[c]);
  }
}

// ---------------- BN finalize: per-channel scale/shift ----------------
__global__ void k_bn_finalize(const float* __restrict__ stats, float invN,
                              const float* __restrict__ g, const float* __restrict__ bb,
                              float* __restrict__ ss, int C) {
  int c = threadIdx.x;
  if (c >= C) return;
  float mean = stats[c] * invN;
  float var  = stats[C + c] * invN - mean * mean;
  float r    = rsqrtf(var + EPSF_);
  float sc   = g[c] * r;
  ss[c]      = sc;
  ss[C + c]  = bb[c] - mean * sc;
}

// normalize+relu previous layer output into next layer's perm-layout A [M,32]
__global__ void k_prep(const _Float16* __restrict__ Y, int ldy, int Cin,
                       const float* __restrict__ ss, _Float16* __restrict__ A, int M) {
  int t = blockIdx.x * blockDim.x + threadIdx.x;
  if (t >= M * 32) return;
  int m = t >> 5, kp = t & 31;
  int k = permk(kp);           // logical channel stored at this physical slot
  float v = 0.f;
  if (k < Cin) v = fmaxf((float)Y[(size_t)m * ldy + k] * ss[k] + ss[Cin + k], 0.f);
  A[t] = (_Float16)v;
}

// BN+relu then max over ns samples -> point features into fusA (perm layout)
__global__ void k_maxpool(const _Float16* __restrict__ Y, int ldy, int ns, int C,
                          const float* __restrict__ ss, _Float16* __restrict__ fusA, int off) {
  int t = blockIdx.x * blockDim.x + threadIdx.x;
  if (t >= MS_ * C) return;
  int m = t / C, c = t % C;
  float sc = ss[c], sh = ss[C + c];
  float acc = 0.f;   // relu lower bound
  for (int j = 0; j < ns; ++j) {
    float v = (float)Y[((size_t)m * ns + j) * ldy + c] * sc + sh;
    acc = fmaxf(acc, v);
  }
  fusA[(size_t)m * 192 + permk(off + c)] = (_Float16)acc;
}

// ---------------- bilinear grid sample -> imgA f16 [MS,32] (perm layout) ----------------
__global__ void k_grid_sample(const float* __restrict__ image, const float* __restrict__ li_xy,
                              _Float16* __restrict__ imgA) {
  int t = blockIdx.x * blockDim.x + threadIdx.x;
  if (t >= MS_) return;
  int b = t / S_;
  const float* fm = image + (size_t)b * CIMG_ * HIMG_ * WIMG_;
  float x = (li_xy[t * 2 + 0] + 1.f) * (WIMG_ * 0.5f) - 0.5f;
  float y = (li_xy[t * 2 + 1] + 1.f) * (HIMG_ * 0.5f) - 0.5f;
  float x0 = floorf(x), y0 = floorf(y);
  float wx = x - x0, wy = y - y0;
  float acc[CIMG_];
#pragma unroll
  for (int c = 0; c < CIMG_; ++c) acc[c] = 0.f;
  for (int tp = 0; tp < 4; ++tp) {
    float xi = x0 + (float)(tp & 1);
    float yi = y0 + (float)(tp >> 1);
    float w = ((tp & 1) ? wx : 1.f - wx) * ((tp >> 1) ? wy : 1.f - wy);
    bool valid = (xi >= 0.f) && (xi <= (float)(WIMG_ - 1)) &&
                 (yi >= 0.f) && (yi <= (float)(HIMG_ - 1));
    int xc = (int)fminf(fmaxf(xi, 0.f), (float)(WIMG_ - 1));
    int yc = (int)fminf(fmaxf(yi, 0.f), (float)(HIMG_ - 1));
    float wv = valid ? w : 0.f;
    const float* base = fm + (size_t)yc * WIMG_ + xc;
#pragma unroll
    for (int c = 0; c < CIMG_; ++c) acc[c] += base[(size_t)c * HIMG_ * WIMG_] * wv;
  }
#pragma unroll
  for (int c = 0; c < CIMG_; ++c) imgA[(size_t)t * 32 + permk(c)] = (_Float16)acc[c];
}

// attention: sigmoid(tanh(ri+rp) @ fc3_w + fc3_b)
__global__ void k_att(const _Float16* __restrict__ Yri, const _Float16* __restrict__ Yrp,
                      const float* __restrict__ fc1_b, const float* __restrict__ fc2_b,
                      const float* __restrict__ fc3_w, const float* __restrict__ fc3_b,
                      float* __restrict__ att) {
  int t = blockIdx.x * blockDim.x + threadIdx.x;
  if (t >= MS_) return;
  float dot = 0.f;
#pragma unroll
  for (int c = 0; c < RC_; ++c) {
    float v = (float)Yri[(size_t)t * 32 + c] + fc1_b[c] +
              (float)Yrp[(size_t)t * 32 + c] + fc2_b[c];
    dot += tanhf(v) * fc3_w[c];
  }
  float z = dot + fc3_b[0];
  att[t] = 1.f / (1.f + __expf(-z));
}

// img_new = relu(bn(ia_conv)) * att -> fusA cols [96,192) (perm layout)
__global__ void k_ia_post(const _Float16* __restrict__ Yia, const float* __restrict__ ss,
                          const float* __restrict__ att, _Float16* __restrict__ fusA) {
  int t = blockIdx.x * blockDim.x + threadIdx.x;
  if (t >= MS_ * PC_) return;
  int m = t / PC_, p = t % PC_;
  float v = (float)Yia[(size_t)m * PC_ + p] * ss[p] + ss[PC_ + p];
  v = fmaxf(v, 0.f) * att[m];
  fusA[(size_t)m * 192 + permk(PC_ + p)] = (_Float16)v;
}

// final: relu(bn(fus)) -> out[b, o, s]
__global__ void k_out(const _Float16* __restrict__ Yfus, const float* __restrict__ ss,
                      float* __restrict__ out) {
  int t = blockIdx.x * blockDim.x + threadIdx.x;
  if (t >= MS_ * PC_) return;
  int m = t / PC_, o = t % PC_;
  float v = fmaxf((float)Yfus[(size_t)m * PC_ + o] * ss[o] + ss[PC_ + o], 0.f);
  int b = m / S_, s = m % S_;
  out[((size_t)b * PC_ + o) * S_ + s] = v;
}

// ---------------- host driver ----------------
extern "C" void kernel_launch(void* const* d_in, const int* in_sizes, int n_in,
                              void* d_out, int out_size, void* d_ws, size_t ws_size,
                              hipStream_t stream) {
  (void)in_sizes; (void)out_size; (void)ws_size;
  if (n_in < 35) return;
  const float* xyz   = (const float*)d_in[0];
  const float* feats = (const float*)d_in[1];
  const float* image = (const float*)d_in[2];
  const float* xy    = (const float*)d_in[3];
  auto P = [&](int i) { return (const float*)d_in[i]; };

  uint8_t* ws = (uint8_t*)d_ws;
  size_t off = 0;
  auto alloc = [&](size_t bytes) -> void* {
    void* p = ws + off;
    off += (bytes + 255) & ~(size_t)255;
    return p;
  };

  float*    dists  = (float*)alloc((size_t)B_ * N_ * 4);
  int*      fpsidx = (int*)alloc((size_t)B_ * S_ * 4);
  float*    newxyz = (float*)alloc((size_t)B_ * S_ * 3 * 4);
  float*    lixy   = (float*)alloc((size_t)B_ * S_ * 2 * 4);
  int*      idx0   = (int*)alloc((size_t)MS_ * NS0_ * 4);
  int*      idx1   = (int*)alloc((size_t)MS_ * NS1_ * 4);
  _Float16* A0     = (_Float16*)alloc((size_t)M0_ * 32 * 2);
  _Float16* A1     = (_Float16*)alloc((size_t)M1_ * 32 * 2);
  _Float16* Y0     = (_Float16*)alloc((size_t)M0_ * 64 * 2);
  _Float16* Y1     = (_Float16*)alloc((size_t)M1_ * 64 * 2);
  _Float16* imgA   = (_Float16*)alloc((size_t)MS_ * 32 * 2);
  _Float16* Yri    = (_Float16*)alloc((size_t)MS_ * 32 * 2);
  _Float16* Yrp    = (_Float16*)alloc((size_t)MS_ * 32 * 2);
  _Float16* Yia    = (_Float16*)alloc((size_t)MS_ * 96 * 2);
  _Float16* fusA   = (_Float16*)alloc((size_t)MS_ * 192 * 2);
  _Float16* Yfus   = (_Float16*)alloc((size_t)MS_ * 96 * 2);
  float*    attb   = (float*)alloc((size_t)MS_ * 4);
  float*    stats  = (float*)alloc(192 * 4);
  float*    ssb    = (float*)alloc(192 * 4);
  _Float16* w00p   = (_Float16*)alloc(32 * 16 * 2);
  _Float16* w01p   = (_Float16*)alloc(32 * 16 * 2);
  _Float16* w02p   = (_Float16*)alloc(32 * 32 * 2);
  _Float16* w10p   = (_Float16*)alloc(32 * 32 * 2);
  _Float16* w11p   = (_Float16*)alloc(32 * 32 * 2);
  _Float16* w12p   = (_Float16*)alloc(32 * 64 * 2);
  _Float16* fc1p   = (_Float16*)alloc(32 * 32 * 2);
  _Float16* fc2p   = (_Float16*)alloc(96 * 32 * 2);
  _Float16* iap    = (_Float16*)alloc(32 * 96 * 2);
  _Float16* fusp   = (_Float16*)alloc(192 * 96 * 2);

  auto pack = [&](const float* src, _Float16* dst, int K, int Nc, int KT, int Np) {
    int n = KT * Np * 32;
    k_pack_w<<<(n + 255) / 256, 256, 0, stream>>>(src, dst, K, Nc, KT, Np);
  };
  pack(P(4),  w00p, 9, 16, 1, 16);
  pack(P(7),  w01p, 16, 16, 1, 16);
  pack(P(10), w02p, 16, 32, 1, 32);
  pack(P(13), w10p, 9, 32, 1, 32);
  pack(P(16), w11p, 32, 32, 1, 32);
  pack(P(19), w12p, 32, 64, 1, 64);
  pack(P(25), fc1p, 32, 24, 1, 32);
  pack(P(27), fc2p, 96, 24, 3, 32);
  pack(P(22), iap, 32, 96, 1, 96);
  pack(P(31), fusp, 192, 96, 6, 96);

  k_fps<<<B_, 1024, 0, stream>>>(xyz, dists, fpsidx);
  k_gather<<<(MS_ + 255) / 256, 256, 0, stream>>>(xyz, xy, fpsidx, newxyz, lixy);
  k_ballquery<<<(MS_ + 255) / 256, 256, 0, stream>>>(xyz, newxyz, idx0, idx1);
  k_group<<<(M0_ + 255) / 256, 256, 0, stream>>>(xyz, feats, newxyz, idx0, NS0_, A0, M0_);
  k_group<<<(M1_ + 255) / 256, 256, 0, stream>>>(xyz, feats, newxyz, idx1, NS1_, A1, M1_);

  auto gblocks = [&](int M) { return (M / 16 + GEMM_WAVES * GEMM_MT - 1) / (GEMM_WAVES * GEMM_MT); };
  auto zstat = [&]() { k_zero_f32<<<1, 192, 0, stream>>>(stats, 192); };
  auto fin = [&](float invN, const float* g, const float* bb, int C) {
    k_bn_finalize<<<1, 128, 0, stream>>>(stats, invN, g, bb, ssb, C);
  };

  // ----- scale 0 MLP chain (M0 rows, ns=16) -----
  zstat();
  k_gemm_t<1, 1><<<gblocks(M0_), 256, 0, stream>>>(A0, 32, w00p, Y0, 64, M0_, stats);
  fin(1.f / M0_, P(5), P(6), 16);
  k_prep<<<(M0_ * 32 + 255) / 256, 256, 0, stream>>>(Y0, 64, 16, ssb, A0, M0_);
  zstat();
  k_gemm_t<1, 1><<<gblocks(M0_), 256, 0, stream>>>(A0, 32, w01p, Y0, 64, M0_, stats);
  fin(1.f / M0_, P(8), P(9), 16);
  k_prep<<<(M0_ * 32 + 255) / 256, 256, 0, stream>>>(Y0, 64, 16, ssb, A0, M0_);
  zstat();
  k_gemm_t<1, 2><<<gblocks(M0_), 256, 0, stream>>>(A0, 32, w02p, Y0, 64, M0_, stats);
  fin(1.f / M0_, P(11), P(12), 32);
  k_maxpool<<<(MS_ * 32 + 255) / 256, 256, 0, stream>>>(Y0, 64, NS0_, 32, ssb, fusA, 0);

  // ----- scale 1 MLP chain (M1 rows, ns=32) -----
  zstat();
  k_gemm_t<1, 2><<<gblocks(M1_), 256, 0, stream>>>(A1, 32, w10p, Y1, 64, M1_, stats);
  fin(1.f / M1_, P(14), P(15), 32);
  k_prep<<<(M1_ * 32 + 255) / 256, 256, 0, stream>>>(Y1, 64, 32, ssb, A1, M1_);
  zstat();
  k_gemm_t<1, 2><<<gblocks(M1_), 256, 0, stream>>>(A1, 32, w11p, Y1, 64, M1_, stats);
  fin(1.f / M1_, P(17), P(18), 32);
  k_prep<<<(M1_ * 32 + 255) / 256, 256, 0, stream>>>(Y1, 64, 32, ssb, A1, M1_);
  zstat();
  k_gemm_t<1, 4><<<gblocks(M1_), 256, 0, stream>>>(A1, 32, w12p, Y1, 64, M1_, stats);
  fin(1.f / M1_, P(20), P(21), 64);
  k_maxpool<<<(MS_ * 64 + 255) / 256, 256, 0, stream>>>(Y1, 64, NS1_, 64, ssb, fusA, 32);

  // ----- image branch -----
  k_grid_sample<<<(MS_ + 255) / 256, 256, 0, stream>>>(image, lixy, imgA);
  k_gemm_t<1, 2><<<gblocks(MS_), 256, 0, stream>>>(imgA, 32, fc1p, Yri, 32, MS_, nullptr);
  k_gemm_t<3, 2><<<gblocks(MS_), 256, 0, stream>>>(fusA, 192, fc2p, Yrp, 32, MS_, nullptr);
  k_att<<<(MS_ + 255) / 256, 256, 0, stream>>>(Yri, Yrp, P(26), P(28), P(29), P(30), attb);
  zstat();
  k_gemm_t<1, 6><<<gblocks(MS_), 256, 0, stream>>>(imgA, 32, iap, Yia, 96, MS_, stats);
  fin(1.f / MS_, P(23), P(24), 96);
  k_ia_post<<<(MS_ * 96 + 255) / 256, 256, 0, stream>>>(Yia, ssb, attb, fusA);

  // ----- fusion + final BN/relu (fus_b cancels inside BN mean subtraction) -----
  zstat();
  k_gemm_t<6, 6><<<gblocks(MS_), 256, 0, stream>>>(fusA, 192, fusp, Yfus, 96, MS_, stats);
  fin(1.f / MS_, P(33), P(34), 96);
  k_out<<<(MS_ * 96 + 255) / 256, 256, 0, stream>>>(Yfus, ssb, (float*)d_out);
}
